// Model_39676907881454
// MI455X (gfx1250) — matrix operand order repaired and checked by reading, not compiled, not run on hardware
//
#include <hip/hip_runtime.h>

typedef __attribute__((ext_vector_type(16))) _Float16 v16h;
typedef __attribute__((ext_vector_type(2)))  _Float16 h2;
typedef __attribute__((ext_vector_type(8)))  float    v8f;

#define S_LEN 2048
#define HDIM  32
#define NHEAD 64            // B*H = 4*16
#define QTILES (S_LEN / 16) // 128
#define SCALE  0.1767766952966369f                       // 1/sqrt(32)
#define LOG2E  1.4426950408889634f
#define SCALE2 (0.1767766952966369f * 1.4426950408889634f)
#define MSHIFT 8.0f          // constant softmax shift (cancels in O/l)

static __device__ __forceinline__ v8f wmma_f16(v16h a, v16h b, v8f c) {
  return __builtin_amdgcn_wmma_f32_16x16x32_f16(
      false, a, false, b, (short)0, c, false, false);
}

// swap the two 16-lane halves of the wave (identity lane mapping)
static __device__ __forceinline__ int xhalf(int x) {
  return __builtin_amdgcn_permlanex16(x, x, 0x76543210, 0xfedcba98, false, false);
}

// ---------------------------------------------------------------------------
// Pre-pass: K (f32,[head][key][d]) -> Kh (f16, same layout)
//           V (f32,[head][key][d]) -> Vt (f16,[head][d][key])
//           Mask (f32,[q][k])      -> M2 = Mask*log2(e) - MSHIFT   (f32)
// ---------------------------------------------------------------------------
__global__ __launch_bounds__(256) void
cvt_kernel(const float* __restrict__ K, const float* __restrict__ V,
           const float* __restrict__ Mask, _Float16* __restrict__ Kh,
           _Float16* __restrict__ Vt, float* __restrict__ M2) {
  const size_t N = (size_t)NHEAD * S_LEN * HDIM;
  const size_t i = (size_t)blockIdx.x * 256 + threadIdx.x;
  if (i < N) {
    Kh[i] = (_Float16)K[i];
  } else if (i < 2 * N) {
    const size_t j = i - N;                 // Vt flat index: [head][d][key]
    const int key  = (int)(j & (S_LEN - 1));
    const int d    = (int)((j >> 11) & (HDIM - 1));
    const int hd   = (int)(j >> 16);
    Vt[j] = (_Float16)V[(size_t)hd * S_LEN * HDIM + (size_t)key * HDIM + d];
  } else {
    const size_t j = i - 2 * N;             // mask index (S*S elements)
    M2[j] = Mask[j] * LOG2E - MSHIFT;
  }
}

// ---------------------------------------------------------------------------
// Fast path: S^T = K @ Q^T, O^T = V^T @ P^T.  No LDS, no running max,
// denominators as per-lane fp32 sums (one cross-half add at the end).
// 4 WMMAs per 32-key block = the minimum for this tiling.
// ---------------------------------------------------------------------------
__global__ __launch_bounds__(128) void
fattn_f16_kernel(const float* __restrict__ Q, const _Float16* __restrict__ Kh,
                 const _Float16* __restrict__ Vt, const float* __restrict__ M2,
                 float* __restrict__ Out) {
  const int lane = threadIdx.x & 31;
  const int wave = threadIdx.x >> 5;
  const int idx  = blockIdx.x * 4 + wave;
  const int head = idx >> 7;
  const int qt   = idx & (QTILES - 1);
  const int q_base = qt * 16;

  const float*    q  = Q   + (size_t)head * S_LEN * HDIM;
  const _Float16* kh = Kh  + (size_t)head * S_LEN * HDIM;
  const _Float16* vt = Vt  + (size_t)head * HDIM * S_LEN;
  float*          o  = Out + (size_t)head * S_LEN * HDIM;

  const int hh = lane >> 4;   // half-wave id
  const int ln = lane & 15;

  // Q^T as B operand (K=d 32, N=q): lane q = q_base+ln, d = hh*8+e / 16+hh*8+e
  v16h b_q;
  {
    const float* qr = q + (size_t)(q_base + ln) * HDIM + hh * 8;
#pragma unroll
    for (int e = 0; e < 8; ++e) { b_q[e] = (_Float16)qr[e]; b_q[e + 8] = (_Float16)qr[16 + e]; }
  }

  v8f oT0 = {}, oT1 = {};   // O^T accumulators: d = hh*8+r and 16+hh*8+r
  float l_part = 0.0f;      // this lane's half of the softmax denominator

  const float* m2row = M2 + (size_t)(q_base + ln) * S_LEN;  // this lane's q row

  for (int kb = 0; kb < S_LEN; kb += 32) {
    // K A tiles (M=key, K=d): row key = kb(+16) + ln; two 8-half chunks of d
    v16h a_k0, a_k1;
    {
      const _Float16* kr0 = kh + (size_t)(kb + ln) * HDIM + hh * 8;
      const _Float16* kr1 = kr0 + 16 * HDIM;
#pragma unroll
      for (int e = 0; e < 8; ++e) {
        a_k0[e]     = kr0[e];
        a_k0[e + 8] = kr0[16 + e];
        a_k1[e]     = kr1[e];
        a_k1[e + 8] = kr1[16 + e];
      }
    }
    // V^T A tiles (M=d, K=key): row d = ln (tile0) / 16+ln (tile1)
    v16h a_v0, a_v1;
    {
      const _Float16* vr0 = vt + (size_t)ln * S_LEN + kb + hh * 8;
      const _Float16* vr1 = vr0 + 16 * S_LEN;
#pragma unroll
      for (int e = 0; e < 8; ++e) {
        a_v0[e]     = vr0[e];
        a_v0[e + 8] = vr0[16 + e];
        a_v1[e]     = vr1[e];
        a_v1[e + 8] = vr1[16 + e];
      }
    }
    if (kb + 32 < S_LEN) {
      __builtin_prefetch(kh + (size_t)(kb + 32 + ln) * HDIM, 0, 3);
      __builtin_prefetch(vt + (size_t)ln * S_LEN + kb + 32, 0, 3);
    }

    // S^T tiles: c0 = keys kb+hh*8+r, c1 = keys kb+16+hh*8+r; col q = ln
    v8f c0 = {}, c1 = {};
    c0 = wmma_f16(a_k0, b_q, c0);
    c1 = wmma_f16(a_k1, b_q, c1);

    // p = exp2(qk*scale*log2e + mask*log2e - shift); mask loads contiguous.
    // Each lane's 16 exps cover keys {hh*8..+7} u {16+hh*8..+7} for q = ln:
    // accumulate them into l_part (partner half holds the complement).
    int pk0[4], pk1[4];
#pragma unroll
    for (int j = 0; j < 4; ++j) {
      const float p0a = __builtin_amdgcn_exp2f(c0[2 * j]     * SCALE2 + m2row[kb + hh * 8 + 2 * j]);
      const float p0b = __builtin_amdgcn_exp2f(c0[2 * j + 1] * SCALE2 + m2row[kb + hh * 8 + 2 * j + 1]);
      const float p1a = __builtin_amdgcn_exp2f(c1[2 * j]     * SCALE2 + m2row[kb + 16 + hh * 8 + 2 * j]);
      const float p1b = __builtin_amdgcn_exp2f(c1[2 * j + 1] * SCALE2 + m2row[kb + 16 + hh * 8 + 2 * j + 1]);
      l_part += (p0a + p0b) + (p1a + p1b);
      union { h2 h; int i; } u0, u1;
      u0.h = (h2){(_Float16)p0a, (_Float16)p0b};
      u1.h = (h2){(_Float16)p1a, (_Float16)p1b};
      pk0[j] = u0.i;
      pk1[j] = u1.i;
    }

    // Assemble P^T in B layout (K=key 32, N=q): no lane exchange is needed.
    union { v16h v; int i[8]; } bp;
#pragma unroll
    for (int j = 0; j < 4; ++j) {
      const int x0 = xhalf(pk0[j]);
      const int x1 = xhalf(pk1[j]);
      bp.i[j]     = pk0[j];
      bp.i[j + 4] = pk1[j];
    }

    oT0 = wmma_f16(a_v0, bp.v, oT0);   // O^T rows d = hh*8+r
    oT1 = wmma_f16(a_v1, bp.v, oT1);   // O^T rows d = 16+hh*8+r
  }

  // epilogue: combine the two half-denominators, one reciprocal per lane,
  // contiguous 8-float stores
  const float l_tot = l_part + __int_as_float(xhalf(__float_as_int(l_part)));
  const float inv   = 1.0f / l_tot;
  float* orow = o + (size_t)(q_base + ln) * HDIM + hh * 8;
#pragma unroll
  for (int r = 0; r < 8; ++r) {
    orow[r]      = oT0[r] * inv;
    orow[16 + r] = oT1[r] * inv;
  }
}

// ---------------------------------------------------------------------------
// Fallback (workspace too small): fp32 K/V, online softmax, LDS transpose
// ---------------------------------------------------------------------------
#define DPP_MAX_STAGE(x, ctrl)                                                 \
  (x) = fmaxf((x), __int_as_float(__builtin_amdgcn_update_dpp(                 \
                  0, __float_as_int(x), (ctrl), 0xF, 0xF, true)))

static __device__ __forceinline__ float half16_max(float x) {
  DPP_MAX_STAGE(x, 0xB1);
  DPP_MAX_STAGE(x, 0x4E);
  DPP_MAX_STAGE(x, 0x141);
  DPP_MAX_STAGE(x, 0x140);
  return x;
}

__global__ __launch_bounds__(128) void
fattn_kernel(const float* __restrict__ Q, const float* __restrict__ K,
             const float* __restrict__ V, const float* __restrict__ Mask,
             float* __restrict__ Out) {
  __shared__ _Float16 lds_p[4][16 * 32];

  const int lane = threadIdx.x & 31;
  const int wave = threadIdx.x >> 5;
  const int idx  = blockIdx.x * 4 + wave;
  const int head = idx >> 7;
  const int qt   = idx & (QTILES - 1);
  const int q_base = qt * 16;

  const size_t head_off = (size_t)head * S_LEN * HDIM;
  const float* q = Q + head_off;
  const float* k = K + head_off;
  const float* v = V + head_off;
  float*       o = Out + head_off;

  const int hh = lane >> 4;
  const int ln = lane & 15;

  v16h a_q;
  {
    const float* qr = q + (size_t)(q_base + ln) * HDIM + hh * 16;
#pragma unroll
    for (int e = 0; e < 8; ++e) {
      a_q[e]     = (_Float16)qr[e];
      a_q[e + 8] = (_Float16)qr[8 + e];
    }
  }

  v16h b_ones;
#pragma unroll
  for (int e = 0; e < 16; ++e) b_ones[e] = (_Float16)1.0f;

  v8f o0 = {}, o1 = {};
  v8f l_acc = {};
  float m_r[8];
#pragma unroll
  for (int r = 0; r < 8; ++r) m_r[r] = -__builtin_huge_valf();

  _Float16* pbuf = lds_p[wave];

  for (int kb = 0; kb < S_LEN; kb += 32) {
    const float* kr0 = k + (size_t)(kb + ln) * HDIM + hh * 16;
    const float* kr1 = kr0 + 16 * HDIM;
    v16h bk0, bk1;
#pragma unroll
    for (int e = 0; e < 16; ++e) {
      bk0[e] = (_Float16)kr0[e];
      bk1[e] = (_Float16)kr1[e];
    }

    v8f c0 = {}, c1 = {};
    c0 = wmma_f16(a_q, bk0, c0);
    c1 = wmma_f16(a_q, bk1, c1);

    const float* mrow = Mask + (size_t)(q_base + hh * 8) * S_LEN + kb + ln;
    float s0[8], s1[8];
#pragma unroll
    for (int r = 0; r < 8; ++r) {
      s0[r] = c0[r] * SCALE + mrow[(size_t)r * S_LEN];
      s1[r] = c1[r] * SCALE + mrow[(size_t)r * S_LEN + 16];
    }

    v8f l_s;
#pragma unroll
    for (int r = 0; r < 8; ++r) {
      const float mx   = half16_max(fmaxf(s0[r], s1[r]));
      const float mnew = fmaxf(m_r[r], mx);
      const float corr = __expf(m_r[r] - mnew);
      m_r[r] = mnew;
      const float p0 = __expf(s0[r] - mnew);
      const float p1 = __expf(s1[r] - mnew);
      o0[r]  *= corr;
      o1[r]  *= corr;
      l_s[r]  = l_acc[r] * corr;
      pbuf[(hh * 8 + r) * 32 + ln]      = (_Float16)p0;
      pbuf[(hh * 8 + r) * 32 + 16 + ln] = (_Float16)p1;
    }

    asm volatile("s_wait_dscnt 0" ::: "memory");

    v16h a_p;
    {
      const _Float16* pr = pbuf + ln * 32 + hh * 16;
#pragma unroll
      for (int e = 0; e < 8; ++e) {
        a_p[e]     = pr[e];
        a_p[e + 8] = pr[8 + e];
      }
    }

    l_acc = wmma_f16(a_p, b_ones, l_s);

    const float* vr = v + (size_t)(kb + hh * 16) * HDIM;
    v16h bv0, bv1;
#pragma unroll
    for (int e = 0; e < 16; ++e) {
      bv0[e] = (_Float16)vr[(size_t)e * HDIM + ln];
      bv1[e] = (_Float16)vr[(size_t)e * HDIM + 16 + ln];
    }

    o0 = wmma_f16(a_p, bv0, o0);
    o1 = wmma_f16(a_p, bv1, o1);
  }

#pragma unroll
  for (int r = 0; r < 8; ++r) {
    const float inv = 1.0f / l_acc[r];
    const size_t row = (size_t)(q_base + hh * 8 + r) * HDIM;
    o[row + ln]      = o0[r] * inv;
    o[row + 16 + ln] = o1[r] * inv;
  }
}

extern "C" void kernel_launch(void* const* d_in, const int* in_sizes, int n_in,
                              void* d_out, int out_size, void* d_ws, size_t ws_size,
                              hipStream_t stream) {
  const float* q    = (const float*)d_in[0];
  const float* k    = (const float*)d_in[1];
  const float* v    = (const float*)d_in[2];
  const float* mask = (const float*)d_in[3];
  float* out        = (float*)d_out;

  const size_t nelem = (size_t)NHEAD * S_LEN * HDIM;            // 4.19M
  const size_t mcnt  = (size_t)S_LEN * S_LEN;                   // 4.19M
  const size_t need  = 2 * nelem * sizeof(_Float16) + mcnt * sizeof(float); // 33.6MB

  if (ws_size >= need) {
    _Float16* kh = (_Float16*)d_ws;
    _Float16* vt = kh + nelem;
    float*    m2 = (float*)(vt + nelem);
    const size_t total = 2 * nelem + mcnt;
    cvt_kernel<<<(unsigned)(total / 256), 256, 0, stream>>>(k, v, mask, kh, vt, m2);
    fattn_f16_kernel<<<NHEAD * QTILES / 4, 128, 0, stream>>>(q, kh, vt, m2, out);
  } else {
    fattn_kernel<<<NHEAD * QTILES / 4, 128, 0, stream>>>(q, k, v, mask, out);
  }
}
